// SCAConv_31069793419919
// MI455X (gfx1250) — matrix operand order repaired and checked by reading, not compiled, not run on hardware
//
#include <hip/hip_runtime.h>
#include <hip/hip_bf16.h>

typedef __attribute__((ext_vector_type(16))) _Float16 v16h;
typedef __attribute__((ext_vector_type(8)))  _Float16 v8h;
typedef __attribute__((ext_vector_type(8)))  float    v8f;
typedef __attribute__((ext_vector_type(4)))  float    v4f;

#define IN_C   16
#define OUT_C  32
#define IKK    144          // IN_C*K*K
#define LSP    4096         // 64*64 locations
#define NCOL   4608         // OUT_C*IKK
#define KDIM   32           // HID*2
#define OBLK   589824       // LSP*IKK  (per-o block in flat Y)

// ---------------------------------------------------------------------------
// Phase 1a: h[r][j] = relu(b1[j] + inp(r) . W1[j]) stored as f16, r in [0,4096)
// inp(r) = [xn, 1-xn, yn, 1-yn, c_param(8)], xn=(r/64)/64, yn=(r%64)/64
// (meshgrid(ow,oh,'ij') width-major flatten). Batch-invariant -> computed once.
// ---------------------------------------------------------------------------
__global__ void sca_mlp_h(const float* __restrict__ c_param,
                          const float* __restrict__ W1,
                          const float* __restrict__ b1,
                          _Float16* __restrict__ hbuf) {
  int gid = blockIdx.x * blockDim.x + threadIdx.x;   // 131072 threads
  int r = gid >> 5;
  int j = gid & 31;
  float xn = (float)(r >> 6) * (1.0f / 64.0f);
  float yn = (float)(r & 63) * (1.0f / 64.0f);
  float inp[12];
  inp[0] = xn; inp[1] = 1.0f - xn; inp[2] = yn; inp[3] = 1.0f - yn;
#pragma unroll
  for (int k = 0; k < 8; ++k) inp[4 + k] = c_param[k];
  const float* w = W1 + j * 12;
  float acc = b1[j];
#pragma unroll
  for (int k = 0; k < 12; ++k) acc = fmaf(inp[k], w[k], acc);
  acc = fmaxf(acc, 0.0f);
  hbuf[r * KDIM + j] = (_Float16)acc;
}

// ---------------------------------------------------------------------------
// Phase 1b: Yf = h(4096x32,f16) @ W2^T(32x4608) + b2 + kernel-fold, f32.
// One wave per 16-row stripe x 128-col group; A-frag loaded once, 8 column tiles per wave,
// one v_wmma_f32_16x16x32_f16 per 16x16 tile (K=32 in a single WMMA).
// Epilogue folds  + b2[col] + kernel[row/128][col%144]  so phase 2 is a pure
// dot against Yf.
// ---------------------------------------------------------------------------
__global__ void sca_wmma_y(const _Float16* __restrict__ hbuf,
                           const float* __restrict__ W2,
                           const float* __restrict__ b2,
                           const float* __restrict__ kern,
                           float* __restrict__ Yf) {
  int wave = (blockIdx.x * blockDim.x + threadIdx.x) >> 5;   // 9216 waves
  int lane = threadIdx.x & 31;
  int rowTile  = wave / 36;          // 256 row tiles
  int colGroup = wave % 36;          // 36 groups of 8 col tiles
  int r0 = rowTile << 4;
  int lanelo = lane & 15;
  int hi     = lane >> 4;

  // A fragment (16x32 f16): lane holds row M=lane%16; K-chunks [8*hi,8*hi+8)
  // and [16+8*hi, 16+8*hi+8) per the CDNA5 16-bit A layout.
  const _Float16* hrow = hbuf + (r0 + lanelo) * KDIM;
  v8h ca = *(const v8h*)(hrow + 8 * hi);
  v8h cb = *(const v8h*)(hrow + 16 + 8 * hi);
  v16h afrag;
#pragma unroll
  for (int j = 0; j < 8; ++j) { afrag[j] = ca[j]; afrag[8 + j] = cb[j]; }

  int o = r0 >> 7;                       // row/128 constant across the tile
  const float* krow = kern + o * IKK;

#pragma unroll
  for (int t = 0; t < 8; ++t) {
    int n = colGroup * 128 + t * 16 + lanelo;   // global column (lane's N)
    // B fragment (32x16): lane holds column n, K-chunks [8*hi,8*hi+8) and [16+8*hi,16+8*hi+8),
    // f32 -> f16 converted in flight (W2 is row-major (4608,32) = B^T).
    const float* wrow = W2 + n * KDIM + hi * 8;
    v4f w0 = ((const v4f*)wrow)[0];
    v4f w1 = ((const v4f*)wrow)[1];
    v4f w2 = ((const v4f*)wrow)[4];
    v4f w3 = ((const v4f*)wrow)[5];
    v16h bfrag;
#pragma unroll
    for (int j = 0; j < 4; ++j) {
      bfrag[j]      = (_Float16)w0[j];
      bfrag[4 + j]  = (_Float16)w1[j];
      bfrag[8 + j]  = (_Float16)w2[j];
      bfrag[12 + j] = (_Float16)w3[j];
    }

    v8f acc = {};
    acc = __builtin_amdgcn_wmma_f32_16x16x32_f16(
        /*neg_a=*/false, afrag, /*neg_b=*/false, bfrag,
        /*c_mod=*/(short)0, acc, /*reuse_a=*/false, /*reuse_b=*/false);

    float add = b2[n] + krow[n % IKK];
#pragma unroll
    for (int v = 0; v < 8; ++v) {
      int row = r0 + v + 8 * hi;           // C/D layout: M = v + 8*(lane>=16)
      Yf[(size_t)row * NCOL + n] = acc[v] + add;
    }
  }
}

// ---------------------------------------------------------------------------
// Phase 2: out[b,o,oh,ow] = bias[o] + sum_i xs[i] * Yf[o*589824 + l*144 + i]
// Block = (oh, b); x halo slice staged in LDS; thread does 8 (o,ow) pairs,
// Y read as contiguous float4s (576 B per (o,l)), fully unrolled 144-dot.
// Y (75.5 MB) is resident in the 192 MB L2 across the 8x batch re-reads.
// ---------------------------------------------------------------------------
__global__ void sca_out(const float* __restrict__ x,
                        const float* __restrict__ bias,
                        const float* __restrict__ Yf,
                        float* __restrict__ out) {
  int oh = blockIdx.x;   // 0..63
  int b  = blockIdx.y;   // 0..7
  __shared__ float xs[IN_C * 3 * 66];     // [c][ki][ow+kj], halo cols -1..64
  int tid = threadIdx.x;

  for (int idx = tid; idx < IN_C * 3 * 66; idx += 256) {
    int c   = idx / 198;
    int rem = idx - c * 198;
    int rr  = rem / 66;
    int col = rem - rr * 66;
    int gr = oh - 1 + rr;
    int gc = col - 1;
    float v = 0.0f;
    if (gr >= 0 && gr < 64 && gc >= 0 && gc < 64)
      v = x[(((size_t)b * IN_C + c) * 64 + gr) * 64 + gc];
    xs[idx] = v;
  }
  __syncthreads();

#pragma unroll
  for (int s = 0; s < 8; ++s) {
    int p  = s * 256 + tid;       // 2048 (o,ow) pairs per (b,oh)
    int o  = p >> 6;
    int ow = p & 63;
    int l  = oh * 64 + ow;
    const v4f* yptr = (const v4f*)(Yf + (size_t)o * OBLK + (size_t)l * IKK);
    float acc = bias[o];
#pragma unroll
    for (int t = 0; t < 36; ++t) {
      v4f y = yptr[t];
#pragma unroll
      for (int e = 0; e < 4; ++e) {
        const int i  = t * 4 + e;
        const int c  = i / 9;
        const int ki = (i % 9) / 3;
        const int kj = i % 3;
        acc = fmaf(xs[(c * 3 + ki) * 66 + ow + kj], y[e], acc);
      }
    }
    out[(((size_t)b * OUT_C + o) * 64 + oh) * 64 + ow] = acc;
  }
}

// ---------------------------------------------------------------------------
extern "C" void kernel_launch(void* const* d_in, const int* in_sizes, int n_in,
                              void* d_out, int out_size, void* d_ws, size_t ws_size,
                              hipStream_t stream) {
  const float* x    = (const float*)d_in[0];
  const float* kern = (const float*)d_in[1];
  const float* bias = (const float*)d_in[2];
  const float* cpar = (const float*)d_in[3];
  const float* W1   = (const float*)d_in[4];
  const float* b1   = (const float*)d_in[5];
  const float* W2   = (const float*)d_in[6];
  const float* b2   = (const float*)d_in[7];

  float*    Yf   = (float*)d_ws;                                   // 75.5 MB
  _Float16* hbuf = (_Float16*)((char*)d_ws +
                               (size_t)LSP * NCOL * sizeof(float)); // 256 KB
  float* out = (float*)d_out;

  // Phase 1a: batch-invariant MLP hidden layer (4096x32) -> f16
  sca_mlp_h<<<512, 256, 0, stream>>>(cpar, W1, b1, hbuf);
  // Phase 1b: WMMA GEMM (M=4096,N=4608,K=32) + b2 + kernel fold -> Yf
  sca_wmma_y<<<1152, 256, 0, stream>>>(hbuf, W2, b2, kern, Yf);
  // Phase 2: fused im2col dot against Yf + bias -> out (8,32,64,64)
  sca_out<<<dim3(64, 8), 256, 0, stream>>>(x, bias, Yf, out);
}
